// VSSBlock_1632087572669
// MI455X (gfx1250) — matrix operand order repaired and checked by reading, not compiled, not run on hardware
//
#include <hip/hip_runtime.h>
#include <math.h>

// ---------------------------------------------------------------------------
// VSS block (VMamba SS2D) for MI455X / gfx1250, wave32, WMMA f16->f32.
// B=4 H=W=64 C=128 DI=256 N=16 R=8 K=4, L=4096, M=B*L=16384 rows.
// Weights pre-packed into WMMA B-fragment order (f16, lane-contiguous 32B).
// out_proj GEMM stages its 64KB B panel in LDS via TDM tensor_load_to_lds.
// ---------------------------------------------------------------------------

typedef __attribute__((ext_vector_type(16))) _Float16 v16h;
typedef __attribute__((ext_vector_type(8)))  _Float16 v8h;
typedef __attribute__((ext_vector_type(8)))  float    v8f;
typedef unsigned int u32x4 __attribute__((ext_vector_type(4)));
typedef int          i32x4 __attribute__((ext_vector_type(4)));
typedef int          i32x8 __attribute__((ext_vector_type(8)));

#if defined(__has_builtin)
#if __has_builtin(__builtin_amdgcn_tensor_load_to_lds) && \
    __has_builtin(__builtin_amdgcn_s_wait_tensorcnt)
#define USE_TDM 1
#endif
#endif

#define DEV __device__ __forceinline__

DEV float siluf(float x)     { return x / (1.f + __expf(-x)); }
DEV float softplusf(float x) { return x > 20.f ? x : log1pf(__expf(x)); }
DEV float geluf(float x)     { return 0.5f * x * (1.f + erff(x * 0.70710678118654752f)); }

// ---- WMMA fragment loaders (gfx1250 wave32 16x16x32 f16 layouts) -----------
// A 16x32 (MxK): lane = row M=lane&15, half=lane>>4; elems 0..7 at K=k0+8*half,
// elems 8..15 at K=k0+16+8*half (two aligned 16B chunks -> 2x b128).
DEV v16h load_a_f16(const _Float16* A, int lda, int row0, int k0) {
  int lane = threadIdx.x & 31;
  const _Float16* p = A + (row0 + (lane & 15)) * lda + k0 + 8 * (lane >> 4);
  v8h lo = *(const v8h*)p;
  v8h hi = *(const v8h*)(p + 16);
  return __builtin_shufflevector(lo, hi, 0,1,2,3,4,5,6,7,8,9,10,11,12,13,14,15);
}

// A fragment from f32 source with K bound (dt GEMM: K=8 padded to 32).
DEV v16h load_a_f32(const float* A, int lda, int row0, int k0, int kmax) {
  int lane = threadIdx.x & 31;
  const float* p = A + (row0 + (lane & 15)) * lda;
  int base = k0 + 8 * (lane >> 4);
  v16h a;
#pragma unroll
  for (int e = 0; e < 8; ++e) { int k = base + e;      a[e]     = (_Float16)(k < kmax ? p[k] : 0.f); }
#pragma unroll
  for (int e = 0; e < 8; ++e) { int k = base + 16 + e; a[8 + e] = (_Float16)(k < kmax ? p[k] : 0.f); }
  return a;
}

// Packed B: fragment (kb, tN) for lane = 16 contiguous halves (32B aligned).
DEV v16h load_b_packed(const _Float16* Bp, int numTN, int kb, int tN) {
  int lane = threadIdx.x & 31;
  return *(const v16h*)(Bp + ((((kb * numTN + tN) * 32) + lane) << 4));
}

DEV v8f wmma_step(v16h a, v16h b, v8f c) {
  return __builtin_amdgcn_wmma_f32_16x16x32_f16(false, a, false, b, (short)0, c, false, false);
}

// ---- Weight pre-pack: f32 (strided/transposed) -> f16 WMMA fragment order --
// Logical B element (k, col) = src[k*sk + col*sn]; zero-padded to Kpad/Npad.
__global__ __launch_bounds__(256) void k_pack_b(const float* __restrict__ src,
                                                int sk, int sn, int Klog, int Nlog,
                                                int numTN, int nKblk,
                                                _Float16* __restrict__ dst) {
  int idx = blockIdx.x * 256 + threadIdx.x;
  if (idx >= nKblk * numTN * 512) return;
  int e    = idx & 15;
  int lane = (idx >> 4) & 31;
  int rest = idx >> 9;
  int tN   = rest % numTN;
  int kb   = rest / numTN;
  int k    = kb * 32 + 8 * (lane >> 4) + (e & 7) + 16 * (e >> 3);
  int col  = tN * 16 + (lane & 15);
  float v = (k < Klog && col < Nlog) ? src[k * sk + col * sn] : 0.f;
  dst[idx] = (_Float16)v;
}

// ---- LayerNorm (one wave per row) -> f16 ------------------------------------
template <int CD>
__global__ __launch_bounds__(256) void k_ln(const float* __restrict__ X,
                                            const float* __restrict__ w,
                                            const float* __restrict__ bb,
                                            _Float16* __restrict__ out) {
  int row  = (blockIdx.x * blockDim.x + threadIdx.x) >> 5;
  int lane = threadIdx.x & 31;
  const float* x = X + row * CD;
  float s = 0.f, s2 = 0.f;
  for (int i = lane; i < CD; i += 32) { float v = x[i]; s += v; s2 += v * v; }
  for (int o = 16; o; o >>= 1) { s += __shfl_xor(s, o, 32); s2 += __shfl_xor(s2, o, 32); }
  float mu = s * (1.f / CD);
  float rs = rsqrtf(s2 * (1.f / CD) - mu * mu + 1e-5f);
  for (int i = lane; i < CD; i += 32)
    out[row * CD + i] = (_Float16)(((x[i] - mu) * rs) * w[i] + bb[i]);
}

// ---- GEMM 1: xn[16384,128] @ Wp_in -> xz f32 [16384,512]; 16x64 per wave ----
__global__ __launch_bounds__(256) void k_gemm_inproj(const _Float16* __restrict__ A,
                                                     const _Float16* __restrict__ Bp,
                                                     float* __restrict__ Out) {
  int wv = (blockIdx.x * blockDim.x + threadIdx.x) >> 5;
  int tM = wv >> 3, g = wv & 7;                       // 8 groups of 4 N-tiles
  v8f c0 = {}, c1 = {}, c2 = {}, c3 = {};
#pragma unroll
  for (int kb = 0; kb < 4; ++kb) {
    v16h a = load_a_f16(A, 128, tM * 16, kb * 32);
    c0 = wmma_step(a, load_b_packed(Bp, 32, kb, g * 4 + 0), c0);
    c1 = wmma_step(a, load_b_packed(Bp, 32, kb, g * 4 + 1), c1);
    c2 = wmma_step(a, load_b_packed(Bp, 32, kb, g * 4 + 2), c2);
    c3 = wmma_step(a, load_b_packed(Bp, 32, kb, g * 4 + 3), c3);
  }
  int lane = threadIdx.x & 31, n = lane & 15, hf = lane >> 4;
  float* o = Out + g * 64 + n;
#pragma unroll
  for (int r = 0; r < 8; ++r) {
    int rowoff = (tM * 16 + r + 8 * hf) * 512;
    o[rowoff] = c0[r]; o[rowoff + 16] = c1[r]; o[rowoff + 32] = c2[r]; o[rowoff + 48] = c3[r];
  }
}

// ---- Depthwise 3x3 conv (SAME) + bias + SiLU; x_ = xz[:, :256] --------------
__global__ __launch_bounds__(256) void k_conv(const float* __restrict__ XZ,
                                              const float* __restrict__ CW,
                                              const float* __restrict__ CB,
                                              float* __restrict__ XC,
                                              _Float16* __restrict__ XC16) {
  int i = blockIdx.x * 256 + threadIdx.x;          // (b*L + l) * 256 + d
  int d = i & 255, row = i >> 8;
  int l = row & 4095, b = row >> 12;
  int h = l >> 6, w = l & 63;
  float acc = CB[d];
#pragma unroll
  for (int dy = -1; dy <= 1; ++dy)
#pragma unroll
    for (int dx = -1; dx <= 1; ++dx) {
      int hh = h + dy, ww = w + dx;
      if ((unsigned)hh < 64u && (unsigned)ww < 64u)
        acc += XZ[(b * 4096 + hh * 64 + ww) * 512 + d] * CW[d * 9 + (dy + 1) * 3 + (dx + 1)];
    }
  float sv = siluf(acc);
  XC[i]   = sv;
  XC16[i] = (_Float16)sv;
}

// ---- GEMM 2: per-dir x_proj: xc[16384,256] @ Wp_xp[k] -> dbl[k][16384,48] ---
__global__ __launch_bounds__(256) void k_gemm_xproj(const _Float16* __restrict__ A,
                                                    const _Float16* __restrict__ Bpk,
                                                    float* __restrict__ Dbl) {
  int wv = (blockIdx.x * blockDim.x + threadIdx.x) >> 5;
  int kd = wv >> 10, tM = wv & 1023;
  const _Float16* Bp = Bpk + kd * 12288;            // 8 kblk x 3 tN x 512
  v8f c0 = {}, c1 = {}, c2 = {};
  for (int kb = 0; kb < 8; ++kb) {
    v16h a = load_a_f16(A, 256, tM * 16, kb * 32);
    c0 = wmma_step(a, load_b_packed(Bp, 3, kb, 0), c0);
    c1 = wmma_step(a, load_b_packed(Bp, 3, kb, 1), c1);
    c2 = wmma_step(a, load_b_packed(Bp, 3, kb, 2), c2);
  }
  int lane = threadIdx.x & 31, n = lane & 15, hf = lane >> 4;
  float* o = Dbl + kd * 16384 * 48;
#pragma unroll
  for (int r = 0; r < 8; ++r) {
    int rowoff = (tM * 16 + r + 8 * hf) * 48;
    o[rowoff + n] = c0[r];
    o[rowoff + 16 + n] = c1[r];
    if (n < 8) o[rowoff + 32 + n] = c2[r];           // cols 40..47 are padding
  }
}

// ---- GEMM 3: dts[16384,8] @ Wp_dt[k] + dt_b -> softplus = delta -------------
__global__ __launch_bounds__(256) void k_gemm_dt(const float* __restrict__ Dbl,
                                                 const _Float16* __restrict__ Bpk,
                                                 const float* __restrict__ DtB,
                                                 float* __restrict__ Delta) {
  int wv = (blockIdx.x * blockDim.x + threadIdx.x) >> 5;
  int kd = wv >> 12;
  int rem = wv & 4095;
  int tM = rem >> 2, g = rem & 3;                    // 4 groups of 4 N-tiles
  const float*    Asrc = Dbl + kd * 16384 * 48;      // cols 0..7 are dts
  const _Float16* Bp   = Bpk + kd * 8192;            // 1 kblk x 16 tN x 512
  v16h a = load_a_f32(Asrc, 48, tM * 16, 0, 8);      // K padded 8 -> 32
  v8f c0 = {}, c1 = {}, c2 = {}, c3 = {};
  c0 = wmma_step(a, load_b_packed(Bp, 16, 0, g * 4 + 0), c0);
  c1 = wmma_step(a, load_b_packed(Bp, 16, 0, g * 4 + 1), c1);
  c2 = wmma_step(a, load_b_packed(Bp, 16, 0, g * 4 + 2), c2);
  c3 = wmma_step(a, load_b_packed(Bp, 16, 0, g * 4 + 3), c3);
  int lane = threadIdx.x & 31, n = lane & 15, hf = lane >> 4;
  int col = g * 64 + n;
  const float* bias = DtB + kd * 256 + col;
  float* o = Delta + kd * 16384 * 256 + col;
#pragma unroll
  for (int r = 0; r < 8; ++r) {
    int rowoff = (tM * 16 + r + 8 * hf) * 256;
    o[rowoff]      = softplusf(c0[r] + bias[0]);
    o[rowoff + 16] = softplusf(c1[r] + bias[16]);
    o[rowoff + 32] = softplusf(c2[r] + bias[32]);
    o[rowoff + 48] = softplusf(c3[r] + bias[48]);
  }
}

// ---- y init: merged skip term  y = (sum_k Ds[k,d]) * xc ---------------------
__global__ __launch_bounds__(256) void k_yinit(const float* __restrict__ XC,
                                               const float* __restrict__ Ds,
                                               float* __restrict__ Y) {
  int i = blockIdx.x * 256 + threadIdx.x;
  int d = i & 255;
  Y[i] = XC[i] * (Ds[d] + Ds[256 + d] + Ds[512 + d] + Ds[768 + d]);
}

// ---- Selective scan: one lane per (b,k,d); N=16 state in registers ----------
DEV int dir_map(int k, int t) {
  if (k == 0) return t;                                    // row-major
  if (k == 1) return ((t & 63) << 6) | (t >> 6);           // column-major
  if (k == 2) return 4095 - t;                             // reversed row
  int u = 4095 - t; return ((u & 63) << 6) | (u >> 6);     // reversed col
}

__global__ __launch_bounds__(256) void k_scan(const float* __restrict__ Delta,
                                              const float* __restrict__ Dbl,
                                              const float* __restrict__ XC,
                                              const float* __restrict__ A_log,
                                              float* __restrict__ Y) {
  int tid = blockIdx.x * 256 + threadIdx.x;     // 0..4095
  int d = tid & 255;
  int k = (tid >> 8) & 3;
  int b = tid >> 10;
  const float* dl = Delta + (k * 16384 + b * 4096) * 256 + d;
  const float* db = Dbl   + (k * 16384 + b * 4096) * 48;
  const float* xb = XC    + b * 4096 * 256 + d;
  float*       yb = Y     + b * 4096 * 256 + d;
  float Arow[16], h[16];
#pragma unroll
  for (int n = 0; n < 16; ++n) {
    Arow[n] = -__expf(A_log[(k * 256 + d) * 16 + n]);
    h[n] = 0.f;
  }
  for (int t = 0; t < 4096; ++t) {
    int l = dir_map(k, t);
    int lnext = dir_map(k, t < 4095 ? t + 1 : t);
    __builtin_prefetch(db + lnext * 48, 0, 1);           // global_prefetch_b8
    __builtin_prefetch(dl + lnext * 256, 0, 1);
    float dt = dl[l * 256];
    float xt = xb[l * 256];
    const float* bc = db + l * 48;                        // B at 8..23, C at 24..39
    float dx = dt * xt;
    float yv = 0.f;
#pragma unroll
    for (int n = 0; n < 16; ++n) {
      float ea = __expf(dt * Arow[n]);
      h[n] = ea * h[n] + dx * bc[8 + n];
      yv += h[n] * bc[24 + n];
    }
    atomicAdd(&yb[l * 256], yv);                          // cross-merge sum
  }
}

// ---- out-norm LN + z-gate -> f16 --------------------------------------------
__global__ __launch_bounds__(256) void k_gate(const float* __restrict__ Y,
                                              const float* __restrict__ XZ,
                                              const float* __restrict__ w,
                                              const float* __restrict__ bb,
                                              _Float16* __restrict__ out) {
  int row  = (blockIdx.x * blockDim.x + threadIdx.x) >> 5;
  int lane = threadIdx.x & 31;
  const float* y = Y + row * 256;
  float s = 0.f, s2 = 0.f;
  for (int i = lane; i < 256; i += 32) { float v = y[i]; s += v; s2 += v * v; }
  for (int o = 16; o; o >>= 1) { s += __shfl_xor(s, o, 32); s2 += __shfl_xor(s2, o, 32); }
  float mu = s * (1.f / 256), rs = rsqrtf(s2 * (1.f / 256) - mu * mu + 1e-5f);
  for (int i = lane; i < 256; i += 32) {
    float yn = ((y[i] - mu) * rs) * w[i] + bb[i];
    float z  = XZ[row * 512 + 256 + i];
    out[row * 256 + i] = (_Float16)(yn * siluf(z));
  }
}

// ---- GEMM 4: yg[16384,256] @ Wp_out + input -> xres.  B panel (64KB) staged
// in LDS via TDM tensor_load_to_lds; fragments come from ds_load_b128. --------
__global__ __launch_bounds__(256) void k_gemm_outproj(const _Float16* __restrict__ A,
                                                      const _Float16* __restrict__ Bp,
                                                      const float* __restrict__ X0,
                                                      float* __restrict__ XRes) {
  __shared__ _Float16 sB[32768] __attribute__((aligned(32)));  // 8 kblk x 8 tN x 512
#ifdef USE_TDM
  if ((threadIdx.x >> 5) == 0) {                   // one wave issues the TDM op
    unsigned long long ga = (unsigned long long)(uintptr_t)Bp;
    unsigned lds = (unsigned)(uintptr_t)&sB[0];    // ADDR[31:0] = LDS byte offset
    u32x4 g0;
    g0[0] = 1u;                                            // count=1, flags 0
    g0[1] = lds;                                           // lds_addr
    g0[2] = (unsigned)(ga & 0xFFFFFFFFu);                  // global_addr[31:0]
    g0[3] = (unsigned)((ga >> 32) & 0x1FFFFFFu) | 0x80000000u; // [56:32] | type=2
    const unsigned elems = 8192;                           // 64KB as 8-byte elems
    i32x8 g1;
    g1[0] = (int)(3u << 16);                               // data_size = 8B
    g1[1] = (int)((elems & 0xFFFFu) << 16);                // tensor_dim0[15:0]
    g1[2] = (int)((elems >> 16) & 0xFFFFu) | (1 << 16);    // dim0 hi | tensor_dim1=1
    g1[3] = (int)((elems & 0xFFFFu) << 16);                // tile_dim0 = 8192
    g1[4] = 1;                                             // tile_dim1 = 1
    g1[5] = (int)elems;                                    // tensor_dim0_stride
    g1[6] = 0; g1[7] = 0;
    i32x4 z4 = {0, 0, 0, 0};
    i32x8 z8 = {0, 0, 0, 0, 0, 0, 0, 0};
    // amdgpu-toolchain (clang-23) 6-arg form: (g0, g1, g2, g3, g4, cpol)
    __builtin_amdgcn_tensor_load_to_lds(g0, g1, z4, z4, z8, 0);
    __builtin_amdgcn_s_wait_tensorcnt((short)0);
  }
  __syncthreads();
#else
  {
    const uint4* src = (const uint4*)Bp;
    uint4* dst = (uint4*)sB;
    for (int i = threadIdx.x; i < 4096; i += 256) dst[i] = src[i];
    __syncthreads();
  }
#endif
  int tM = blockIdx.x * 8 + (threadIdx.x >> 5);    // 8 waves: 8 M-tiles/block
  v8f acc[8];
#pragma unroll
  for (int j = 0; j < 8; ++j) acc[j] = (v8f){};
  for (int kb = 0; kb < 8; ++kb) {
    v16h a = load_a_f16(A, 256, tM * 16, kb * 32);
#pragma unroll
    for (int j = 0; j < 8; ++j)
      acc[j] = wmma_step(a, load_b_packed(sB, 8, kb, j), acc[j]);
  }
  int lane = threadIdx.x & 31, n = lane & 15, hf = lane >> 4;
#pragma unroll
  for (int j = 0; j < 8; ++j)
#pragma unroll
    for (int r = 0; r < 8; ++r) {
      int idx = (tM * 16 + r + 8 * hf) * 128 + j * 16 + n;
      XRes[idx] = X0[idx] + acc[j][r];
    }
}

// ---- GEMM 5: h2n[16384,128] @ Wp_fc1 + b -> GELU -> f16 ---------------------
__global__ __launch_bounds__(256) void k_gemm_fc1(const _Float16* __restrict__ A,
                                                  const _Float16* __restrict__ Bp,
                                                  const float* __restrict__ bias,
                                                  _Float16* __restrict__ Out) {
  int wv = (blockIdx.x * blockDim.x + threadIdx.x) >> 5;
  int tM = wv >> 3, g = wv & 7;
  v8f c0 = {}, c1 = {}, c2 = {}, c3 = {};
#pragma unroll
  for (int kb = 0; kb < 4; ++kb) {
    v16h a = load_a_f16(A, 128, tM * 16, kb * 32);
    c0 = wmma_step(a, load_b_packed(Bp, 32, kb, g * 4 + 0), c0);
    c1 = wmma_step(a, load_b_packed(Bp, 32, kb, g * 4 + 1), c1);
    c2 = wmma_step(a, load_b_packed(Bp, 32, kb, g * 4 + 2), c2);
    c3 = wmma_step(a, load_b_packed(Bp, 32, kb, g * 4 + 3), c3);
  }
  int lane = threadIdx.x & 31, n = lane & 15, hf = lane >> 4;
  int col = g * 64 + n;
  const float* bv = bias + col;
  _Float16* o = Out + col;
#pragma unroll
  for (int r = 0; r < 8; ++r) {
    int rowoff = (tM * 16 + r + 8 * hf) * 512;
    o[rowoff]      = (_Float16)geluf(c0[r] + bv[0]);
    o[rowoff + 16] = (_Float16)geluf(c1[r] + bv[16]);
    o[rowoff + 32] = (_Float16)geluf(c2[r] + bv[32]);
    o[rowoff + 48] = (_Float16)geluf(c3[r] + bv[48]);
  }
}

// ---- GEMM 6: h1[16384,512] @ Wp_fc2 + b + xres -> d_out ---------------------
__global__ __launch_bounds__(256) void k_gemm_fc2(const _Float16* __restrict__ A,
                                                  const _Float16* __restrict__ Bp,
                                                  const float* __restrict__ bias,
                                                  const float* __restrict__ XRes,
                                                  float* __restrict__ Out) {
  int tM = blockIdx.x * 8 + (threadIdx.x >> 5);
  v8f acc[8];
#pragma unroll
  for (int j = 0; j < 8; ++j) acc[j] = (v8f){};
  for (int kb = 0; kb < 16; ++kb) {
    v16h a = load_a_f16(A, 512, tM * 16, kb * 32);
#pragma unroll
    for (int j = 0; j < 8; ++j)
      acc[j] = wmma_step(a, load_b_packed(Bp, 8, kb, j), acc[j]);
  }
  int lane = threadIdx.x & 31, n = lane & 15, hf = lane >> 4;
#pragma unroll
  for (int j = 0; j < 8; ++j) {
    float bv = bias[j * 16 + n];
#pragma unroll
    for (int r = 0; r < 8; ++r) {
      int idx = (tM * 16 + r + 8 * hf) * 128 + j * 16 + n;
      Out[idx] = acc[j][r] + bv + XRes[idx];
    }
  }
}

// ---------------------------------------------------------------------------
extern "C" void kernel_launch(void* const* d_in, const int* in_sizes, int n_in,
                              void* d_out, int out_size, void* d_ws, size_t ws_size,
                              hipStream_t stream) {
  const float* input     = (const float*)d_in[0];
  const float* ln1_w     = (const float*)d_in[1];
  const float* ln1_b     = (const float*)d_in[2];
  const float* in_proj_w = (const float*)d_in[3];
  const float* conv_w    = (const float*)d_in[4];
  const float* conv_b    = (const float*)d_in[5];
  const float* x_proj_w  = (const float*)d_in[6];
  const float* dt_w      = (const float*)d_in[7];
  const float* dt_b      = (const float*)d_in[8];
  const float* A_log     = (const float*)d_in[9];
  const float* Ds        = (const float*)d_in[10];
  const float* onorm_w   = (const float*)d_in[11];
  const float* onorm_b   = (const float*)d_in[12];
  const float* out_proj_w= (const float*)d_in[13];
  const float* ln2_w     = (const float*)d_in[14];
  const float* ln2_b     = (const float*)d_in[15];
  const float* fc1_w     = (const float*)d_in[16];
  const float* fc1_b     = (const float*)d_in[17];
  const float* fc2_w     = (const float*)d_in[18];
  const float* fc2_b     = (const float*)d_in[19];
  float* out = (float*)d_out;
  char*  ws  = (char*)d_ws;

  // Scratch layout (bytes); safe reuse across producer/consumer boundaries.
  _Float16* xn16  = (_Float16*)(ws + 0);                        //  4MB (reused: h2n16)
  float*    xz    = (float*)   (ws + (size_t)4  * (1u << 20));  // 32MB
  float*    xc    = (float*)   (ws + (size_t)36 * (1u << 20));  // 16MB
  _Float16* xc16  = (_Float16*)(ws + (size_t)52 * (1u << 20));  //  8MB
  float*    dbl   = (float*)   (ws + (size_t)60 * (1u << 20));  // 12MB (reused: xres)
  float*    delta = (float*)   (ws + (size_t)72 * (1u << 20));  // 64MB (reused: h1_16)
  float*    y     = (float*)   (ws + (size_t)136* (1u << 20));  // 16MB
  _Float16* yg16  = (_Float16*)(ws + (size_t)152* (1u << 20));  //  8MB
  float*    xres  = (float*)   (ws + (size_t)60 * (1u << 20));  // overlaps dbl
  _Float16* h2n16 = (_Float16*)(ws + 0);                        // overlaps xn16
  _Float16* h1_16 = (_Float16*)(ws + (size_t)72 * (1u << 20));  // overlaps delta
  // Packed f16 weights (~0.6MB total)
  _Float16* Wpool = (_Float16*)(ws + (size_t)160 * (1u << 20));
  _Float16* Wp_in  = Wpool;                 // 4 kblk x 32 tN  = 65536
  _Float16* Wp_xp  = Wpool + 65536;         // 4 dirs x (8x3x512) = 49152
  _Float16* Wp_dt  = Wpool + 114688;        // 4 dirs x (1x16x512) = 32768
  _Float16* Wp_out = Wpool + 147456;        // 8 kblk x 8 tN   = 32768
  _Float16* Wp_fc1 = Wpool + 180224;        // 4 kblk x 32 tN  = 65536
  _Float16* Wp_fc2 = Wpool + 245760;        // 16 kblk x 8 tN  = 65536

  dim3 blk(256);
  // -- weight pre-pack (tiny; once per launch) --
  k_pack_b<<<256, blk, 0, stream>>>(in_proj_w, 512, 1, 128, 512, 32, 4, Wp_in);
  for (int kd = 0; kd < 4; ++kd)
    k_pack_b<<<48, blk, 0, stream>>>(x_proj_w + kd * 10240, 1, 256, 256, 40, 3, 8,
                                     Wp_xp + kd * 12288);
  for (int kd = 0; kd < 4; ++kd)
    k_pack_b<<<32, blk, 0, stream>>>(dt_w + kd * 2048, 1, 8, 8, 256, 16, 1,
                                     Wp_dt + kd * 8192);
  k_pack_b<<<128, blk, 0, stream>>>(out_proj_w, 128, 1, 256, 128, 8, 8, Wp_out);
  k_pack_b<<<256, blk, 0, stream>>>(fc1_w, 512, 1, 128, 512, 32, 4, Wp_fc1);
  k_pack_b<<<256, blk, 0, stream>>>(fc2_w, 128, 1, 512, 128, 8, 16, Wp_fc2);

  // -- main pipeline --
  k_ln<128>     <<< 2048, blk, 0, stream>>>(input, ln1_w, ln1_b, xn16);
  k_gemm_inproj <<< 1024, blk, 0, stream>>>(xn16, Wp_in, xz);
  k_conv        <<<16384, blk, 0, stream>>>(xz, conv_w, conv_b, xc, xc16);
  k_gemm_xproj  <<<  512, blk, 0, stream>>>(xc16, Wp_xp, dbl);
  k_gemm_dt     <<< 2048, blk, 0, stream>>>(dbl, Wp_dt, dt_b, delta);
  k_yinit       <<<16384, blk, 0, stream>>>(xc, Ds, y);
  k_scan        <<<   16, blk, 0, stream>>>(delta, dbl, xc, A_log, y);
  k_gate        <<< 2048, blk, 0, stream>>>(y, xz, onorm_w, onorm_b, yg16);
  k_gemm_outproj<<<  128, blk, 0, stream>>>(yg16, Wp_out, input, xres);
  k_ln<128>     <<< 2048, blk, 0, stream>>>(xres, ln2_w, ln2_b, h2n16);
  k_gemm_fc1    <<< 1024, blk, 0, stream>>>(h2n16, Wp_fc1, fc1_b, h1_16);
  k_gemm_fc2    <<<  128, blk, 0, stream>>>(h1_16, Wp_fc2, fc2_b, xres, out);

  (void)in_sizes; (void)n_in; (void)out_size; (void)ws_size;
}
